// MultiHeadAttentionKeras_85650237817140
// MI455X (gfx1250) — matrix operand order repaired and checked by reading, not compiled, not run on hardware
//
#include <hip/hip_runtime.h>

#define F_DIM  2048
#define B_DIM  2
#define H_DIM  1024
#define N_HEADS 16
#define E_DIM  64
#define D3     (3*H_DIM)      // 3072
#define FB     (F_DIM*B_DIM)  // 4096

typedef __attribute__((ext_vector_type(16))) __bf16         v16bf;
typedef __attribute__((ext_vector_type(8)))  float          v8f;
typedef __attribute__((ext_vector_type(16))) unsigned short v16u;
typedef __attribute__((ext_vector_type(8)))  unsigned short v8u;

__device__ __forceinline__ unsigned short f32_to_bf16(float f){
  unsigned int u = __float_as_uint(f);
  u += 0x7FFFu + ((u >> 16) & 1u);     // round-to-nearest-even
  return (unsigned short)(u >> 16);
}

// A-matrix 16x32 bf16 fragment (ISA layout): lane half holds K = {h*8..h*8+7, 16+h*8..16+h*8+7}
// p must point at (row_base + kk + half*8); second chunk at +16 elements. 16B aligned.
__device__ __forceinline__ v16bf load_frag_a(const unsigned short* p){
  v8u lo = *(const v8u*)p;
  v8u hi = *(const v8u*)(p + 16);
  v16u v;
#pragma unroll
  for (int i = 0; i < 8; ++i){ v[i] = lo[i]; v[i+8] = hi[i]; }
  return __builtin_bit_cast(v16bf, v);
}

// B-matrix 32x16 bf16 fragment (ISA layout): lane half holds K = {h*8..h*8+7, 16+h*8..16+h*8+7}
// p must point at (col_base + kk + half*8); second chunk at +16 elements. 16B aligned.
__device__ __forceinline__ v16bf load_frag_b(const unsigned short* p){
  v8u lo = *(const v8u*)p;
  v8u hi = *(const v8u*)(p + 16);
  v16u v;
#pragma unroll
  for (int i = 0; i < 8; ++i){ v[i] = lo[i]; v[i+8] = hi[i]; }
  return __builtin_bit_cast(v16bf, v);
}

__device__ __forceinline__ v8f wmma_bf16(v16bf a, v16bf b, v8f c){
  // (neg_a, A, neg_b, B, c_mod, C, reuse_a, reuse_b)
  return __builtin_amdgcn_wmma_f32_16x16x32_bf16(false, a, false, b, (short)0, c, false, false);
}

// CDNA5 async copy: per-lane 16B global -> LDS, tracked by ASYNCcnt.
// LDS byte offset = low 32 bits of the generic pointer.
__device__ __forceinline__ void async_load_b128(const void* gaddr, void* lds_ptr){
  unsigned lds_off = (unsigned)(size_t)lds_ptr;
  unsigned long long ga = (unsigned long long)(size_t)gaddr;
  asm volatile("global_load_async_to_lds_b128 %0, %1, off"
               :: "v"(lds_off), "v"(ga) : "memory");
}
__device__ __forceinline__ void wait_async0(){
  asm volatile("s_wait_asynccnt 0" ::: "memory");
}

// ---------------------------------------------------------------------------
// fp32 -> bf16 conversion
__global__ void to_bf16_kernel(const float* __restrict__ in,
                               unsigned short* __restrict__ out, int n){
  int i = blockIdx.x*256 + threadIdx.x;
  if (i < n) out[i] = f32_to_bf16(in[i]);
}

// in [R][C] fp32 row-major -> out [C][R] bf16 (transposed, coalesced writes)
__global__ void transpose_to_bf16_kernel(const float* __restrict__ in,
                                         unsigned short* __restrict__ out,
                                         int R, int C){
  int i = blockIdx.x*256 + threadIdx.x;
  if (i < R*C){
    int c = i / R, r = i - c*R;
    out[i] = f32_to_bf16(in[(size_t)r*C + c]);
  }
}

__global__ void copy_tail_kernel(const float* __restrict__ b_out,
                                 float* __restrict__ dst){
  int i = threadIdx.x;
  if (i < H_DIM) dst[i] = b_out[i];
}

// ---------------------------------------------------------------------------
// WMMA GEMM: out[M][Ncols] = A[M][K] * Bt[Ncols][K]^T (+bias). 8 waves/block,
// wave tile 32x64, block tile 128x128. M,Ncols multiples of 128; K mult of 32.
template<bool BIAS, bool OUT_BF16>
__global__ __launch_bounds__(256)
void gemm_wmma_kernel(const unsigned short* __restrict__ A,
                      const unsigned short* __restrict__ Bt,
                      const float* __restrict__ bias,
                      void* __restrict__ outv,
                      int M, int Ncols, int K)
{
  const int tid = threadIdx.x, lane = tid & 31, wid = tid >> 5;
  const int wm = wid >> 1, wn = wid & 1;            // waves: 4 (M) x 2 (N)
  const int m0 = blockIdx.y*128 + wm*32;
  const int n0 = blockIdx.x*128 + wn*64;
  const int kh = lane >> 4, lr = lane & 15;

  v8f acc[2][4];
#pragma unroll
  for (int mi = 0; mi < 2; ++mi)
#pragma unroll
    for (int ni = 0; ni < 4; ++ni) acc[mi][ni] = {};

  const unsigned short* Ap = A  + (size_t)(m0 + lr)*K;
  const unsigned short* Bp = Bt + (size_t)(n0 + lr)*K;

  for (int kk = 0; kk < K; kk += 32){
    if (kk + 32 < K){
      __builtin_prefetch(A  + (size_t)(m0 + lane)*K      + kk + 32, 0, 3);
      __builtin_prefetch(Bt + (size_t)(n0 + lane)*K      + kk + 32, 0, 3);
      __builtin_prefetch(Bt + (size_t)(n0 + 32 + lane)*K + kk + 32, 0, 3);
    }
    v16bf a[2], bfr[4];
#pragma unroll
    for (int mi = 0; mi < 2; ++mi)
      a[mi] = load_frag_a(Ap + (size_t)mi*16*K + kk + kh*8);
#pragma unroll
    for (int ni = 0; ni < 4; ++ni)
      bfr[ni] = load_frag_b(Bp + (size_t)ni*16*K + kk + kh*8);
#pragma unroll
    for (int mi = 0; mi < 2; ++mi)
#pragma unroll
      for (int ni = 0; ni < 4; ++ni)
        acc[mi][ni] = wmma_bf16(a[mi], bfr[ni], acc[mi][ni]);
  }

#pragma unroll
  for (int ni = 0; ni < 4; ++ni){
    const int col = n0 + ni*16 + lr;
    const float bb = BIAS ? bias[col] : 0.0f;
#pragma unroll
    for (int mi = 0; mi < 2; ++mi){
#pragma unroll
      for (int g = 0; g < 8; ++g){
        const int row = m0 + mi*16 + g + kh*8;     // C layout: VGPR g, lane half
        float v = acc[mi][ni][g] + bb;
        if (OUT_BF16)
          ((unsigned short*)outv)[(size_t)row*Ncols + col] = f32_to_bf16(v);
        else
          ((float*)outv)[(size_t)row*Ncols + col] = v;
      }
    }
  }
}

// ---------------------------------------------------------------------------
// Flash attention: one block = (b, n, 64 query rows). 4 waves, each owns one
// 16-row M-tile, full E=64 output width, streaming T in 64-column chunks with
// online softmax. V chunk (8KB) and mask chunk (4KB) for the NEXT iteration are
// double-buffered into LDS with CDNA5 async-to-LDS copies (ASYNCcnt) so data
// movement overlaps the WMMA/softmax work of the current chunk.
// QKV layout: [f*B+b][n*192 + {q:0..63, k:64..127, v:128..191}]
__global__ __launch_bounds__(128)
void attn_flash_kernel(const unsigned short* __restrict__ QKV,
                       const unsigned char* __restrict__ mask,   // [B][F][F] bool
                       unsigned short* __restrict__ Ctx)         // [FB][H] bf16
{
  __shared__ alignas(16) unsigned short lds_vraw[2][64*64]; // V chunk [t][e], double buf
  __shared__ alignas(16) unsigned char  lds_m[2][64*64];    // mask chunk [f_local][t_local]
  __shared__ alignas(32) unsigned short lds_v[64*80];       // V^T [e][t], stride 80
  __shared__ alignas(32) unsigned short lds_p[64*80];       // P   [row][t], stride 80

  const int tid  = threadIdx.x;
  const int lane = tid & 31;
  const int w    = tid >> 5;
  const int kh   = lane >> 4;
  const int lr   = lane & 15;
  const int b = blockIdx.z, n = blockIdx.y;
  const int fblk = blockIdx.x*64;
  const int f0 = fblk + w*16;

  const size_t rstride = (size_t)B_DIM * D3;
  const size_t boff = (size_t)b * D3;
  const unsigned short* Qb = QKV + n*192;
  const unsigned short* Kb = Qb + 64;
  const unsigned short* Vb = Qb + 128 + boff;              // per-(b,n) V base
  const unsigned char*  Mb = mask + (size_t)b*F_DIM*F_DIM + (size_t)fblk*F_DIM;

  // ---- async stage of one T chunk (V rows + mask rows) into buffer `buf`
  auto stage_chunk = [&](int t0, int buf){
    // V: 64 rows x 128B  -> 512 x 16B chunks
    for (int c = tid; c < 512; c += 128){
      int tt = c >> 3, e8 = c & 7;
      async_load_b128(Vb + (size_t)(t0 + tt)*rstride + e8*8,
                      &lds_vraw[buf][tt*64 + e8*8]);
    }
    // mask: 64 rows x 64B -> 256 x 16B chunks
    for (int c = tid; c < 256; c += 128){
      int fr = c >> 2, g16 = c & 3;
      async_load_b128(Mb + (size_t)fr*F_DIM + t0 + g16*16,
                      &lds_m[buf][fr*64 + g16*16]);
    }
  };

  v16bf aq[2];
#pragma unroll
  for (int ks = 0; ks < 2; ++ks)
    aq[ks] = load_frag_a(Qb + (size_t)(f0 + lr)*rstride + boff + ks*32 + kh*8);

  v8f O[4];
#pragma unroll
  for (int ni = 0; ni < 4; ++ni) O[ni] = {};
  float m_run[8], l_run[8];
#pragma unroll
  for (int g = 0; g < 8; ++g){ m_run[g] = -1e30f; l_run[g] = 0.0f; }

  stage_chunk(0, 0);                                 // prologue: chunk 0 in flight

  for (int t0 = 0; t0 < F_DIM; t0 += 64){
    const int cur = (t0 >> 6) & 1;
    wait_async0();                                   // this wave's async copies done
    __syncthreads();                                 // publish buf[cur]; retire old readers
    if (t0 + 64 < F_DIM) stage_chunk(t0 + 64, 1 - cur);  // overlap next chunk's DMA

    // in-LDS transpose V [t][e] -> lds_v [e][t]
    for (int idx = tid; idx < 64*64; idx += 128){
      int tt = idx >> 6, e = idx & 63;
      lds_v[e*80 + tt] = lds_vraw[cur][tt*64 + e];
    }
    __syncthreads();

    // S = Q * K^T  (16 x 64 per wave)
    v8f S[4];
#pragma unroll
    for (int ni = 0; ni < 4; ++ni) S[ni] = {};
#pragma unroll
    for (int ks = 0; ks < 2; ++ks){
#pragma unroll
      for (int ni = 0; ni < 4; ++ni){
        v16bf bk = load_frag_b(Kb + (size_t)(t0 + ni*16 + lr)*rstride + boff
                               + ks*32 + kh*8);
        S[ni] = wmma_bf16(aq[ks], bk, S[ni]);
      }
    }

    // scale + mask (mask broadcast over heads; read from LDS-staged chunk)
    const unsigned char* mrow = lds_m[cur];
#pragma unroll
    for (int ni = 0; ni < 4; ++ni){
      const int tc = ni*16 + lr;
#pragma unroll
      for (int g = 0; g < 8; ++g){
        const int frl = w*16 + g + kh*8;             // block-local query row
        float s = S[ni][g] * 0.125f;                 // 1/sqrt(64)
        if (!mrow[frl*64 + tc]) s = -10000.0f;
        S[ni][g] = s;
      }
    }

    // row max (reduce across 16 lanes inside each wave half) + online rescale
    float corr[8];
#pragma unroll
    for (int g = 0; g < 8; ++g){
      float r = S[0][g];
#pragma unroll
      for (int ni = 1; ni < 4; ++ni) r = fmaxf(r, S[ni][g]);
      r = fmaxf(r, __shfl_xor(r, 1, 32));
      r = fmaxf(r, __shfl_xor(r, 2, 32));
      r = fmaxf(r, __shfl_xor(r, 4, 32));
      r = fmaxf(r, __shfl_xor(r, 8, 32));
      float mn = fmaxf(m_run[g], r);
      corr[g] = __expf(m_run[g] - mn);
      m_run[g] = mn;
    }
#pragma unroll
    for (int ni = 0; ni < 4; ++ni)
#pragma unroll
      for (int g = 0; g < 8; ++g)
        S[ni][g] = __expf(S[ni][g] - m_run[g]);
#pragma unroll
    for (int g = 0; g < 8; ++g){
      float s = S[0][g] + S[1][g] + S[2][g] + S[3][g];
      s += __shfl_xor(s, 1, 32);
      s += __shfl_xor(s, 2, 32);
      s += __shfl_xor(s, 4, 32);
      s += __shfl_xor(s, 8, 32);
      l_run[g] = l_run[g]*corr[g] + s;
#pragma unroll
      for (int ni = 0; ni < 4; ++ni) O[ni][g] *= corr[g];
    }

    // P (bf16) -> LDS (C-layout scatter), then reload as A-fragments
#pragma unroll
    for (int ni = 0; ni < 4; ++ni)
#pragma unroll
      for (int g = 0; g < 8; ++g)
        lds_p[(w*16 + g + kh*8)*80 + ni*16 + lr] = f32_to_bf16(S[ni][g]);

    v16bf ap[2];
#pragma unroll
    for (int ks = 0; ks < 2; ++ks)
      ap[ks] = load_frag_a(&lds_p[(w*16 + lr)*80 + ks*32 + kh*8]);
#pragma unroll
    for (int ni = 0; ni < 4; ++ni){
#pragma unroll
      for (int ks = 0; ks < 2; ++ks){
        v16bf bv = load_frag_b(&lds_v[(ni*16 + lr)*80 + ks*32 + kh*8]);
        O[ni] = wmma_bf16(ap[ks], bv, O[ni]);
      }
    }
  }

  // epilogue: normalize, write ctx as [f*B+b][n*64+e] bf16
#pragma unroll
  for (int ni = 0; ni < 4; ++ni){
#pragma unroll
    for (int g = 0; g < 8; ++g){
      const int fr  = f0 + g + kh*8;
      const int col = n*64 + ni*16 + lr;
      float v = O[ni][g] / l_run[g];
      Ctx[((size_t)fr*B_DIM + b)*H_DIM + col] = f32_to_bf16(v);
    }
  }
}

// ---------------------------------------------------------------------------
extern "C" void kernel_launch(void* const* d_in, const int* in_sizes, int n_in,
                              void* d_out, int out_size, void* d_ws, size_t ws_size,
                              hipStream_t stream)
{
  (void)in_sizes; (void)n_in; (void)out_size; (void)ws_size;
  const float*         q_input = (const float*)d_in[0];
  const unsigned char* mask    = (const unsigned char*)d_in[1];  // jnp bool -> 1 byte
  const float*         w_qkv   = (const float*)d_in[2];
  const float*         b_qkv   = (const float*)d_in[3];
  const float*         w_out   = (const float*)d_in[4];
  const float*         b_out   = (const float*)d_in[5];
  float* out = (float*)d_out;

  // workspace: 48 MB of bf16 staging
  unsigned short* Xb    = (unsigned short*)d_ws;                 // [FB][H]
  unsigned short* Wqkvb = Xb    + (size_t)FB*H_DIM;              // [D3][H]   (W^T)
  unsigned short* Woutb = Wqkvb + (size_t)H_DIM*D3;              // [H][H]    (W^T)
  unsigned short* QKVb  = Woutb + (size_t)H_DIM*H_DIM;           // [FB][D3]
  unsigned short* Ctxb  = QKVb  + (size_t)FB*D3;                 // [FB][H]

  to_bf16_kernel<<<(FB*H_DIM)/256, 256, 0, stream>>>(q_input, Xb, FB*H_DIM);
  transpose_to_bf16_kernel<<<(H_DIM*D3)/256, 256, 0, stream>>>(w_qkv, Wqkvb, H_DIM, D3);
  transpose_to_bf16_kernel<<<(H_DIM*H_DIM)/256, 256, 0, stream>>>(w_out, Woutb, H_DIM, H_DIM);

  // QKV projection (+bias), bf16 out
  gemm_wmma_kernel<true, true><<<dim3(D3/128, FB/128), 256, 0, stream>>>(
      Xb, Wqkvb, b_qkv, QKVb, FB, D3, H_DIM);

  // flash attention per (b, n, 64-row block), async double-buffered V/mask
  attn_flash_kernel<<<dim3(F_DIM/64, N_HEADS, B_DIM), 128, 0, stream>>>(QKVb, mask, Ctxb);

  // output projection, f32 straight to d_out (reference adds no bias here)
  gemm_wmma_kernel<false, false><<<dim3(H_DIM/128, FB/128), 256, 0, stream>>>(
      Ctxb, Woutb, nullptr, out, FB, H_DIM, H_DIM);

  // second tuple output: b_out appended after out
  copy_tail_kernel<<<1, 256, 0, stream>>>(b_out, out + (size_t)FB*H_DIM);
}
